// MOELayer_42786464202757
// MI455X (gfx1250) — matrix operand order repaired and checked by reading, not compiled, not run on hardware
//
#include <hip/hip_runtime.h>

// ---------------------------------------------------------------------------
// MoE layer for MI455X (gfx1250, wave32, WMMA).
// S=4096 tokens, M=1024 model, H=4096 hidden, E=8 experts, top-2, C=1024.
// Expert GEMMs (137 GFLOP fp32) are done with v_wmma_f32_16x16x32_bf16 using
// split-bf16 ("bf16x3") compensation: A=Ahi+Alo, B=Bhi+Blo,
// acc += Ahi*Bhi + Ahi*Blo + Alo*Bhi  (fp32 accumulate) -> ~fp32 accuracy at
// 3 bf16 WMMAs per logical fp32 MAC, ~8x the work-per-issue of the f32 WMMA.
// ---------------------------------------------------------------------------

constexpr int TOK = 4096;   // S
constexpr int DM  = 1024;   // M (model dim)
constexpr int DH  = 4096;   // H (hidden dim)
constexpr int NE  = 8;      // E experts
constexpr int CAP = 1024;   // C capacity per expert

typedef __attribute__((ext_vector_type(16))) __bf16 v16bf;
typedef __attribute__((ext_vector_type(8)))  float  v8f;

union FragBF { uint4 q[2]; v16bf v; };

__device__ __forceinline__ unsigned short f2bf(float f) {
  unsigned u = __float_as_uint(f);
  u += 0x7fffu + ((u >> 16) & 1u);          // round-to-nearest-even
  return (unsigned short)(u >> 16);
}
__device__ __forceinline__ float bf2f(unsigned short h) {
  return __uint_as_float((unsigned)h << 16);
}
__device__ __forceinline__ unsigned pack2(unsigned short a, unsigned short b) {
  return (unsigned)a | ((unsigned)b << 16);
}

// ---------------------------------------------------------------------------
// 0) init: slot_map = -1
// ---------------------------------------------------------------------------
__global__ __launch_bounds__(256)
void init_slotmap_kernel(int* __restrict__ slot_map) {
  int i = blockIdx.x * 256 + threadIdx.x;
  if (i < NE * CAP) slot_map[i] = -1;
}

// ---------------------------------------------------------------------------
// 1) gating: logits -> softmax -> top2 -> gates; one wave (32 lanes) per token
// ---------------------------------------------------------------------------
__global__ __launch_bounds__(256)
void gate_kernel(const float* __restrict__ x, const float* __restrict__ wg,
                 int* __restrict__ ids, float* __restrict__ gates,
                 float* __restrict__ scores) {
  const int lane = threadIdx.x & 31;
  const int wv   = threadIdx.x >> 5;
  const int s    = blockIdx.x * 8 + wv;
  const float* xr = x + (size_t)s * DM;

  float acc[8];
#pragma unroll
  for (int e = 0; e < 8; ++e) acc[e] = 0.f;

  for (int m = lane; m < DM; m += 32) {
    float xv = xr[m];
    const float4* wr = (const float4*)(wg + (size_t)m * 8);
    float4 w0 = wr[0], w1 = wr[1];
    acc[0] += xv * w0.x; acc[1] += xv * w0.y; acc[2] += xv * w0.z; acc[3] += xv * w0.w;
    acc[4] += xv * w1.x; acc[5] += xv * w1.y; acc[6] += xv * w1.z; acc[7] += xv * w1.w;
  }
#pragma unroll
  for (int e = 0; e < 8; ++e) {
#pragma unroll
    for (int off = 16; off; off >>= 1) acc[e] += __shfl_xor(acc[e], off, 32);
  }
  // softmax over 8 experts (redundant per-lane; cheap)
  float mx = acc[0];
#pragma unroll
  for (int e = 1; e < 8; ++e) mx = fmaxf(mx, acc[e]);
  float sum = 0.f, sc[8];
#pragma unroll
  for (int e = 0; e < 8; ++e) { sc[e] = __expf(acc[e] - mx); sum += sc[e]; }
  float inv = 1.f / sum;
#pragma unroll
  for (int e = 0; e < 8; ++e) sc[e] *= inv;

  // top-2, lowest index wins ties (matches lax.top_k)
  int i0 = 0; float v0 = sc[0];
#pragma unroll
  for (int e = 1; e < 8; ++e) { if (sc[e] > v0) { v0 = sc[e]; i0 = e; } }
  int i1 = -1; float v1 = -1.f;
#pragma unroll
  for (int e = 0; e < 8; ++e) { if (e != i0 && sc[e] > v1) { v1 = sc[e]; i1 = e; } }

  float denom = fmaxf(v0 + v1, 1e-7f);
  if (lane == 0) {
    ids[s * 2 + 0]   = i0;
    ids[s * 2 + 1]   = i1;
    gates[s * 2 + 0] = v0 / denom;
    gates[s * 2 + 1] = v1 / denom;
#pragma unroll
    for (int e = 0; e < 8; ++e) scores[(size_t)s * 8 + e] = sc[e];
  }
}

// ---------------------------------------------------------------------------
// 2) capacity assignment: exact sequential semantics via block prefix scans.
//    Single workgroup of 1024 threads; pass k=0 over all tokens, then k=1.
// ---------------------------------------------------------------------------
__global__ __launch_bounds__(1024)
void scan_kernel(const int* __restrict__ ids, int* __restrict__ loc,
                 int* __restrict__ slot_map) {
  __shared__ unsigned sc[1024];
  __shared__ int bases[8];
  const int t = threadIdx.x;
  if (t < 8) bases[t] = 0;
  __syncthreads();

  for (int k = 0; k < 2; ++k) {
    for (int chunk = 0; chunk < TOK / 1024; ++chunk) {
      const int s  = chunk * 1024 + t;
      const int id = ids[s * 2 + k];
      int myloc = -1;
      for (int e = 0; e < 8; ++e) {
        unsigned p = (id == e) ? 1u : 0u;
        sc[t] = p;
        __syncthreads();
        unsigned v = p;
        for (int off = 1; off < 1024; off <<= 1) {
          unsigned add = (t >= off) ? sc[t - off] : 0u;
          __syncthreads();
          v += add;
          sc[t] = v;
          __syncthreads();
        }
        if (p) myloc = bases[e] + (int)v - 1;   // exclusive rank + base
        __syncthreads();
        if (t == 1023) bases[e] += (int)v;      // total count this chunk
        __syncthreads();
      }
      loc[s * 2 + k] = myloc;
      if (myloc >= 0 && myloc < CAP) slot_map[id * CAP + myloc] = s;
    }
  }
}

// ---------------------------------------------------------------------------
// 3+4) expert GEMM: Out[e] (CAP x N) = act( A (CAP x K) @ B[e] (K x N) + b[e] )
//   GATHER: A rows gathered from x via slot_map (empty slot -> zero row)
//   Block tile 128x128, k-chunk 32; 8 waves of 64x32; bf16 hi/lo in LDS.
//   LDS layout stores packed bf16 k-pairs so each 16x16x32 fragment is two
//   ds_load_b128 per lane (stride 20 dwords: 16B aligned, conflict-free).
// ---------------------------------------------------------------------------
template <bool GATHER, bool RELU>
__global__ __launch_bounds__(256)
void moe_gemm_kernel(const float* __restrict__ Abase,    // GATHER: x[S,K] ; else [E*CAP,K]
                     const float* __restrict__ Bbase,    // [E,K,N]
                     const float* __restrict__ biasBase, // [E,N]
                     float* __restrict__ Outbase,        // [E*CAP,N]
                     const int* __restrict__ slot_map,   // [E*CAP]
                     int Kdim, int N) {
  constexpr int BM = 128, BN = 128, BK = 32, SROW = 20;   // 16 k-pairs + pad
  __shared__ unsigned smem[4 * BM * SROW];                 // 40 KB
  unsigned* sAhi = smem;
  unsigned* sAlo = smem + 1 * BM * SROW;
  unsigned* sBhi = smem + 2 * BM * SROW;
  unsigned* sBlo = smem + 3 * BM * SROW;

  const int e    = blockIdx.z;
  const int bx   = blockIdx.x;           // N tile
  const int by   = blockIdx.y;           // row tile (capacity)
  const int tid  = threadIdx.x;
  const int lane = tid & 31, wave = tid >> 5;
  const int wm   = wave & 1, wn = wave >> 1;   // 2 (rows) x 4 (cols) waves
  const int hi16 = lane >> 4, lm = lane & 15;

  const float* __restrict__ B    = Bbase + (size_t)e * Kdim * N + (size_t)bx * BN;
  const float* __restrict__ bias = biasBase + (size_t)e * N + (size_t)bx * BN;
  float* __restrict__ Out        = Outbase + (size_t)e * CAP * N;

  // Per-thread A row pointers: rows tid>>3 + i*32, float4 slot tid&7
  const int qa = tid & 7;
  const float* aptr[4];
#pragma unroll
  for (int i = 0; i < 4; ++i) {
    const int row = by * BM + (tid >> 3) + i * 32;
    if constexpr (GATHER) {
      const int tok = slot_map[e * CAP + row];
      aptr[i] = (tok >= 0) ? (Abase + (size_t)tok * Kdim + qa * 4) : nullptr;
    } else {
      aptr[i] = Abase + ((size_t)e * CAP + row) * Kdim + qa * 4;
    }
  }
  const int bn  = tid & 127;   // B column
  const int kp0 = tid >> 7;    // 0..1

  v8f acc[4][2];
#pragma unroll
  for (int t = 0; t < 4; ++t)
#pragma unroll
    for (int u = 0; u < 2; ++u)
#pragma unroll
      for (int r = 0; r < 8; ++r) acc[t][u][r] = 0.f;

  float4 av[4];
  float  bv0[8], bv1[8];

  auto load_regs = [&](int k0) {
#pragma unroll
    for (int i = 0; i < 4; ++i) {
      if constexpr (GATHER) {
        av[i] = aptr[i] ? *(const float4*)(aptr[i] + k0) : make_float4(0.f, 0.f, 0.f, 0.f);
      } else {
        av[i] = *(const float4*)(aptr[i] + k0);
      }
    }
#pragma unroll
    for (int i = 0; i < 8; ++i) {
      const int kp = kp0 + i * 2;
      const float* p = B + (size_t)(k0 + 2 * kp) * N + bn;
      bv0[i] = p[0];
      bv1[i] = p[N];
    }
  };

  auto stage = [&]() {
#pragma unroll
    for (int i = 0; i < 4; ++i) {
      const int r = (tid >> 3) + i * 32;
      float4 a = av[i];
      unsigned short h0 = f2bf(a.x), h1 = f2bf(a.y), h2 = f2bf(a.z), h3 = f2bf(a.w);
      unsigned short l0 = f2bf(a.x - bf2f(h0)), l1 = f2bf(a.y - bf2f(h1));
      unsigned short l2 = f2bf(a.z - bf2f(h2)), l3 = f2bf(a.w - bf2f(h3));
      sAhi[r * SROW + qa * 2 + 0] = pack2(h0, h1);
      sAhi[r * SROW + qa * 2 + 1] = pack2(h2, h3);
      sAlo[r * SROW + qa * 2 + 0] = pack2(l0, l1);
      sAlo[r * SROW + qa * 2 + 1] = pack2(l2, l3);
    }
#pragma unroll
    for (int i = 0; i < 8; ++i) {
      const int kp = kp0 + i * 2;
      unsigned short h0 = f2bf(bv0[i]), h1 = f2bf(bv1[i]);
      unsigned short l0 = f2bf(bv0[i] - bf2f(h0)), l1 = f2bf(bv1[i] - bf2f(h1));
      sBhi[bn * SROW + kp] = pack2(h0, h1);
      sBlo[bn * SROW + kp] = pack2(l0, l1);
    }
  };

  auto compute = [&]() {
    FragBF bhi[2], blo[2];
    const int bq0 = hi16 ? 8 : 0;   // lanes 16-31 hold K=16..31 pairs
#pragma unroll
    for (int u = 0; u < 2; ++u) {
      const int row = (wn * 32 + u * 16 + lm) * SROW;
      bhi[u].q[0] = *(const uint4*)&sBhi[row + bq0];
      bhi[u].q[1] = *(const uint4*)&sBhi[row + bq0 + 4];
      blo[u].q[0] = *(const uint4*)&sBlo[row + bq0];
      blo[u].q[1] = *(const uint4*)&sBlo[row + bq0 + 4];
    }
    const int aq0 = hi16 ? 8 : 0;   // lanes 16-31 hold K=16..31 pairs
#pragma unroll
    for (int t = 0; t < 4; ++t) {
      FragBF ahi, alo;
      const int row = (wm * 64 + t * 16 + lm) * SROW;
      ahi.q[0] = *(const uint4*)&sAhi[row + aq0];
      ahi.q[1] = *(const uint4*)&sAhi[row + aq0 + 4];
      alo.q[0] = *(const uint4*)&sAlo[row + aq0];
      alo.q[1] = *(const uint4*)&sAlo[row + aq0 + 4];
#pragma unroll
      for (int u = 0; u < 2; ++u) {
        acc[t][u] = __builtin_amdgcn_wmma_f32_16x16x32_bf16(
            false, ahi.v, false, bhi[u].v, (short)0, acc[t][u], false, false);
        acc[t][u] = __builtin_amdgcn_wmma_f32_16x16x32_bf16(
            false, ahi.v, false, blo[u].v, (short)0, acc[t][u], false, false);
        acc[t][u] = __builtin_amdgcn_wmma_f32_16x16x32_bf16(
            false, alo.v, false, bhi[u].v, (short)0, acc[t][u], false, false);
      }
    }
  };

  const int nChunks = Kdim / BK;
  load_regs(0);
  stage();
  __syncthreads();
  for (int c = 0; c < nChunks; ++c) {
    const bool hasNext = (c + 1 < nChunks);
    if (hasNext) load_regs((c + 1) * BK);   // global loads overlap compute
    compute();
    __syncthreads();
    if (hasNext) { stage(); __syncthreads(); }
  }

  // epilogue: bias (+ReLU) and store
#pragma unroll
  for (int t = 0; t < 4; ++t) {
#pragma unroll
    for (int u = 0; u < 2; ++u) {
      const int col = bx * BN + wn * 32 + u * 16 + lm;
      const float b = bias[wn * 32 + u * 16 + lm];
      const int rbase = by * BM + wm * 64 + t * 16 + (hi16 ? 8 : 0);
#pragma unroll
      for (int r = 0; r < 8; ++r) {
        float v = acc[t][u][r] + b;
        if (RELU) v = fmaxf(v, 0.f);
        Out[(size_t)(rbase + r) * N + col] = v;
      }
    }
  }
}

// ---------------------------------------------------------------------------
// 5) combine: out[s] = sum_k gate_k * yo[expert_k*CAP + loc_k]
// ---------------------------------------------------------------------------
__global__ __launch_bounds__(256)
void combine_kernel(const float* __restrict__ yo, const int* __restrict__ ids,
                    const float* __restrict__ gates, const int* __restrict__ loc,
                    float* __restrict__ out) {
  const int s = blockIdx.x;
  const int t = threadIdx.x;
  const int i0 = ids[s * 2 + 0], i1 = ids[s * 2 + 1];
  const int l0 = loc[s * 2 + 0], l1 = loc[s * 2 + 1];
  float g0 = gates[s * 2 + 0], g1 = gates[s * 2 + 1];
  g0 = (l0 >= 0 && l0 < CAP) ? g0 : 0.f;
  g1 = (l1 >= 0 && l1 < CAP) ? g1 : 0.f;
  const int c0 = min(max(l0, 0), CAP - 1);
  const int c1 = min(max(l1, 0), CAP - 1);
  const float4* r0 = (const float4*)(yo + ((size_t)i0 * CAP + c0) * DM);
  const float4* r1 = (const float4*)(yo + ((size_t)i1 * CAP + c1) * DM);
  float4 a = r0[t], b = r1[t], o;
  o.x = g0 * a.x + g1 * b.x;
  o.y = g0 * a.y + g1 * b.y;
  o.z = g0 * a.z + g1 * b.z;
  o.w = g0 * a.w + g1 * b.w;
  ((float4*)(out + (size_t)s * DM))[t] = o;
}

// ---------------------------------------------------------------------------
// 6) l_aux = E * sum_e mean(scores[:,e]) * mean(top1==e)  (deterministic tree)
// ---------------------------------------------------------------------------
__global__ __launch_bounds__(256)
void aux_kernel(const float* __restrict__ scores, const int* __restrict__ ids,
                float* __restrict__ laux_out) {
  __shared__ float wsum[8];
  __shared__ float ME[8], CE[8];
  const int tid = threadIdx.x, lane = tid & 31, wv = tid >> 5;
  float me[8], ce[8];
#pragma unroll
  for (int e = 0; e < 8; ++e) { me[e] = 0.f; ce[e] = 0.f; }
  for (int s = tid; s < TOK; s += 256) {
#pragma unroll
    for (int e = 0; e < 8; ++e) me[e] += scores[(size_t)s * 8 + e];
    const int top1 = ids[s * 2];
#pragma unroll
    for (int e = 0; e < 8; ++e) ce[e] += (top1 == e) ? 1.f : 0.f;
  }
  for (int e = 0; e < 8; ++e) {
    float v = me[e];
#pragma unroll
    for (int off = 16; off; off >>= 1) v += __shfl_xor(v, off, 32);
    if (lane == 0) wsum[wv] = v;
    __syncthreads();
    if (tid == 0) { float s2 = 0.f; for (int w = 0; w < 8; ++w) s2 += wsum[w]; ME[e] = s2; }
    __syncthreads();
    v = ce[e];
#pragma unroll
    for (int off = 16; off; off >>= 1) v += __shfl_xor(v, off, 32);
    if (lane == 0) wsum[wv] = v;
    __syncthreads();
    if (tid == 0) { float s2 = 0.f; for (int w = 0; w < 8; ++w) s2 += wsum[w]; CE[e] = s2; }
    __syncthreads();
  }
  if (tid == 0) {
    float l = 0.f;
    for (int e = 0; e < 8; ++e) l += (ME[e] / (float)TOK) * (CE[e] / (float)TOK);
    laux_out[0] = l * (float)NE;
  }
}

// ---------------------------------------------------------------------------
// launch
// ---------------------------------------------------------------------------
extern "C" void kernel_launch(void* const* d_in, const int* in_sizes, int n_in,
                              void* d_out, int out_size, void* d_ws, size_t ws_size,
                              hipStream_t stream) {
  (void)in_sizes; (void)n_in; (void)out_size; (void)ws_size;
  const float* x    = (const float*)d_in[0];
  const float* wg   = (const float*)d_in[1];
  const float* fc1w = (const float*)d_in[2];
  const float* fc1b = (const float*)d_in[3];
  const float* fc2w = (const float*)d_in[4];
  const float* fc2b = (const float*)d_in[5];
  float* out = (float*)d_out;

  char* ws = (char*)d_ws;
  size_t off = 0;
  auto walloc = [&](size_t bytes) -> void* {
    size_t a = (off + 255) & ~(size_t)255;
    off = a + bytes;
    return (void*)(ws + a);
  };
  int*   slot_map = (int*)  walloc((size_t)NE * CAP * sizeof(int));
  int*   ids      = (int*)  walloc((size_t)TOK * 2 * sizeof(int));
  float* gates    = (float*)walloc((size_t)TOK * 2 * sizeof(float));
  int*   loc      = (int*)  walloc((size_t)TOK * 2 * sizeof(int));
  float* scores   = (float*)walloc((size_t)TOK * 8 * sizeof(float));
  float* hbuf     = (float*)walloc((size_t)NE * CAP * DH * sizeof(float));  // 128 MB
  float* yobuf    = (float*)walloc((size_t)NE * CAP * DM * sizeof(float));  //  32 MB

  init_slotmap_kernel<<<(NE * CAP) / 256, 256, 0, stream>>>(slot_map);
  gate_kernel<<<TOK / 8, 256, 0, stream>>>(x, wg, ids, gates, scores);
  scan_kernel<<<1, 1024, 0, stream>>>(ids, loc, slot_map);

  // fc1: (CAP x M) @ (M x H) + b, ReLU, A gathered from x via slot_map
  moe_gemm_kernel<true, true><<<dim3(DH / 128, CAP / 128, NE), 256, 0, stream>>>(
      x, fc1w, fc1b, hbuf, slot_map, DM, DH);
  // fc2: (CAP x H) @ (H x M) + b
  moe_gemm_kernel<false, false><<<dim3(DM / 128, CAP / 128, NE), 256, 0, stream>>>(
      hbuf, fc2w, fc2b, yobuf, nullptr, DH, DM);

  combine_kernel<<<TOK, 256, 0, stream>>>(yobuf, ids, gates, loc, out);
  aux_kernel<<<1, 256, 0, stream>>>(scores, ids, out + (size_t)TOK * DM);
}
